// LinearInt8_67018669686910
// MI455X (gfx1250) — compile-verified, matrix operand order checked by reading, not run on hardware
//
#include <hip/hip_runtime.h>

// GEMM: C[M,N] = A[M,K] (f32 -> f16) * W[N,K]^T (int8 -> f16), f32 accumulate.
// M = 8192 (B*S), N = 4096, K = 4096.

typedef __attribute__((ext_vector_type(16))) _Float16 v16h;
typedef __attribute__((ext_vector_type(8)))  _Float16 h8;
typedef __attribute__((ext_vector_type(4)))  _Float16 h4;
typedef __attribute__((ext_vector_type(2)))  _Float16 h2;
typedef __attribute__((ext_vector_type(8)))  float    v8f;

#define GK 4096
#define GN 4096
#define BK 32          // K step per stage
#define LDS_STRIDE 40  // 32 halves + 8 pad halves -> conflict-free b128 reads

// f32 pair -> packed f16 (v_cvt_pk_f16_f32, RTZ)
__device__ __forceinline__ h2 pk_f16(float x, float y) {
  return __builtin_bit_cast(h2, __builtin_amdgcn_cvt_pkrtz(x, y));
}

// 4 packed int8 -> 4 f16, exact. u = b ^ 0x80 in [0,255];
// 0x6400|u encodes f16(1152 + b); subtract exact 1152 -> b.
__device__ __forceinline__ h4 cvt_s8x4_to_h4(unsigned w) {
  const unsigned u = w ^ 0x80808080u;
  const unsigned lo = __builtin_amdgcn_perm(u, 0x64646464u, 0x01050004u);
  const unsigned hi = __builtin_amdgcn_perm(u, 0x64646464u, 0x03070206u);
  const h2 bias = {(_Float16)-1152.0f, (_Float16)-1152.0f};
  const h2 l = __builtin_bit_cast(h2, lo) + bias;
  const h2 h = __builtin_bit_cast(h2, hi) + bias;
  h4 r;
  r[0] = l[0]; r[1] = l[1]; r[2] = h[0]; r[3] = h[1];
  return r;
}

__global__ __launch_bounds__(256) void LinearInt8_wmma_kernel(
    const float* __restrict__ A, const signed char* __restrict__ W,
    float* __restrict__ C) {
  // 2 buffers * 128 rows * 40 halves * 2B = 20 KB each for A and B tiles
  __shared__ _Float16 Ash[2][128 * LDS_STRIDE];
  __shared__ _Float16 Bsh[2][128 * LDS_STRIDE];

  const int tid  = threadIdx.x;
  const int lane = tid & 31;
  const int wave = tid >> 5;
  const int wm   = wave & 1;   // 2 waves along M  -> 64 rows each
  const int wn   = wave >> 1;  // 4 waves along N  -> 32 cols each
  const int l    = lane & 15;
  const int half = lane >> 4;  // 0: K 0-7/16-23, 1: K 8-15/24-31 (ISA layout)
  const int kfrag = half * 8;

  const int m0 = blockIdx.y * 128;  // workgroup M base
  const int n0 = blockIdx.x * 128;  // workgroup N base

  // ---- staging register addresses (constant per thread) ----
  // A tile: 128 rows x 32 f32.  4 passes, 32 rows/pass, 8 threads/row x 4 f32.
  const int a_row = tid >> 3;          // 0..31 (+32 per pass)
  const int a_c4  = (tid & 7) * 4;     // 0,4,..,28
  // W tile: 128 rows x 32 i8.  2 threads/row x 16 bytes.
  const int b_row = tid >> 1;          // 0..127
  const int b_seg = (tid & 1) * 16;    // 0 or 16

  float4 aReg[4];
  int4   bReg;

  auto gload = [&](int kk) {
#pragma unroll
    for (int p = 0; p < 4; ++p) {
      const float* pa =
          A + (size_t)(m0 + p * 32 + a_row) * GK + kk + a_c4;
      aReg[p] = *(const float4*)pa;
    }
    const signed char* pb = W + (size_t)(n0 + b_row) * GK + kk + b_seg;
    bReg = *(const int4*)pb;
  };

  auto sstore = [&](int buf) {
#pragma unroll
    for (int p = 0; p < 4; ++p) {
      const h2 p0 = pk_f16(aReg[p].x, aReg[p].y);
      const h2 p1 = pk_f16(aReg[p].z, aReg[p].w);
      h4 h;
      h[0] = p0[0]; h[1] = p0[1]; h[2] = p1[0]; h[3] = p1[1];
      *(h4*)&Ash[buf][(p * 32 + a_row) * LDS_STRIDE + a_c4] = h;
    }
    h8 lo, hi;
    const h4 c0 = cvt_s8x4_to_h4((unsigned)bReg.x);
    const h4 c1 = cvt_s8x4_to_h4((unsigned)bReg.y);
    const h4 c2 = cvt_s8x4_to_h4((unsigned)bReg.z);
    const h4 c3 = cvt_s8x4_to_h4((unsigned)bReg.w);
#pragma unroll
    for (int e = 0; e < 4; ++e) {
      lo[e] = c0[e]; lo[e + 4] = c1[e];
      hi[e] = c2[e]; hi[e + 4] = c3[e];
    }
    _Float16* dst = &Bsh[buf][b_row * LDS_STRIDE + b_seg];
    *(h8*)dst = lo;
    *(h8*)(dst + 8) = hi;
  };

  v8f acc[4][2];
#pragma unroll
  for (int i = 0; i < 4; ++i)
#pragma unroll
    for (int j = 0; j < 2; ++j) acc[i][j] = (v8f)(0.0f);

  // prologue: stage K block 0
  gload(0);
  sstore(0);
  __syncthreads();

  int cur = 0;
  for (int kk = 0; kk < GK; kk += BK) {
    const bool has_next = (kk + BK) < GK;
    if (has_next) gload(kk + BK);  // global loads overlap the WMMAs below

    // ---- consume current LDS buffer ----
    v16h aF[4], bF[2];
#pragma unroll
    for (int i = 0; i < 4; ++i) {
      const _Float16* base =
          &Ash[cur][(wm * 64 + i * 16 + l) * LDS_STRIDE + kfrag];
      h8 lo = *(const h8*)base;
      h8 hi = *(const h8*)(base + 16);
#pragma unroll
      for (int e = 0; e < 8; ++e) { aF[i][e] = lo[e]; aF[i][e + 8] = hi[e]; }
    }
#pragma unroll
    for (int j = 0; j < 2; ++j) {
      const _Float16* base =
          &Bsh[cur][(wn * 32 + j * 16 + l) * LDS_STRIDE + kfrag];
      h8 lo = *(const h8*)base;
      h8 hi = *(const h8*)(base + 16);
#pragma unroll
      for (int e = 0; e < 8; ++e) { bF[j][e] = lo[e]; bF[j][e + 8] = hi[e]; }
    }
#pragma unroll
    for (int i = 0; i < 4; ++i)
#pragma unroll
      for (int j = 0; j < 2; ++j)
        acc[i][j] = __builtin_amdgcn_wmma_f32_16x16x32_f16(
            false, aF[i], false, bF[j], (short)0, acc[i][j], false, false);

    if (has_next) sstore(cur ^ 1);
    __syncthreads();
    cur ^= 1;
  }

  // ---- write back: VGPR r holds M = half*8 + r, lane&15 holds N ----
#pragma unroll
  for (int i = 0; i < 4; ++i)
#pragma unroll
    for (int j = 0; j < 2; ++j)
#pragma unroll
      for (int r = 0; r < 8; ++r) {
        const int row = m0 + wm * 64 + i * 16 + half * 8 + r;
        const int col = n0 + wn * 32 + j * 16 + l;
        C[(size_t)row * GN + col] = acc[i][j][r];
      }
}

extern "C" void kernel_launch(void* const* d_in, const int* in_sizes, int n_in,
                              void* d_out, int out_size, void* d_ws, size_t ws_size,
                              hipStream_t stream) {
  const float*       A = (const float*)d_in[0];
  const signed char* W = (const signed char*)d_in[1];
  float*             C = (float*)d_out;
  const int M = in_sizes[0] / GK;  // 8192 = B*S
  dim3 grid(GN / 128, M / 128);    // 32 x 64 workgroups
  LinearInt8_wmma_kernel<<<grid, 256, 0, stream>>>(A, W, C);
}
